// Net_LSTM_cell_1752346657182
// MI455X (gfx1250) — matrix operand order repaired and checked by reading, not compiled, not run on hardware
//
#include <hip/hip_runtime.h>
#include <cstdint>
#include <cstddef>

// ---------------- CDNA5 WMMA types ----------------
typedef __attribute__((ext_vector_type(16))) __bf16       v16bf;
typedef __attribute__((ext_vector_type(8)))  float        v8f;
typedef __attribute__((ext_vector_type(4)))  unsigned int u32x4;

// ---------------- problem constants ----------------
constexpr int kB  = 4096;   // batch
constexpr int kIN = 4096;   // input dim
constexpr int kFS = 4096;   // feature size
constexpr int kS  = 64;     // sequence steps
constexpr int kH  = 512;    // hidden
constexpr int kD  = 64;     // FS / S
constexpr int kG  = 4 * kH; // 2048 gate width

// LDS B-tile geometry: 256 columns x 32 K-elements (bf16), padded stride.
constexpr int kBStride = 40;            // bf16 elems per col (80B, 16B aligned,
                                        // bank stride 20 words -> conflict-free)
constexpr int kBufElems = 256 * kBStride;

union Frag { u32x4 u4[2]; v16bf v; };

__device__ __forceinline__ v8f wmma_bf16(v16bf a, v16bf b, v8f c) {
  return __builtin_amdgcn_wmma_f32_16x16x32_bf16(
      /*neg_a=*/false, a, /*neg_b=*/false, b,
      /*c_mod=*/(short)0, c, /*reuse_a=*/false, /*reuse_b=*/false);
}

// ---- CDNA5 async global->LDS copy (ASYNCcnt) -------------------------------
__device__ __forceinline__ void async_g2l_b128(unsigned ldsOff, const void* g) {
#if __has_builtin(__builtin_amdgcn_global_load_async_to_lds_b128)
  // exact builtin signature differs across toolchains; use asm for portability
  asm volatile("global_load_async_to_lds_b128 %0, %1, off"
               :: "v"(ldsOff), "v"((unsigned long long)(uintptr_t)g) : "memory");
#else
  asm volatile("global_load_async_to_lds_b128 %0, %1, off"
               :: "v"(ldsOff), "v"((unsigned long long)(uintptr_t)g) : "memory");
#endif
}

__device__ __forceinline__ void wait_async0() {
#if __has_builtin(__builtin_amdgcn_s_wait_asynccnt)
  __builtin_amdgcn_s_wait_asynccnt(0);
#else
  asm volatile("s_wait_asynccnt 0x0" ::: "memory");
#endif
}

// A fragment (16x32 bf16, M x K). lanes 0-15: M=lane, K chunks {0-7,16-23};
// lanes 16-31: M=lane-16, K chunks {8-15,24-31}.
__device__ __forceinline__ v16bf load_a_frag(const __bf16* __restrict__ A, int lda,
                                             int kk, int lane) {
  int row  = lane & 15;
  int ksel = lane >> 4;
  const __bf16* p = A + (size_t)row * lda + kk + ksel * 8;
  Frag f;
  f.u4[0] = *reinterpret_cast<const u32x4*>(p);
  f.u4[1] = *reinterpret_cast<const u32x4*>(p + 16);
  return f.v;
}

// B fragment from LDS-staged tile: lane holds col = base + lane&15,
// K chunks {(lane>>4)*8 .. +7, 16+(lane>>4)*8 .. +7} (2x ds_load_b128).
__device__ __forceinline__ v16bf load_b_frag_lds(const __bf16* buf, int lane) {
  int col   = lane & 15;
  int khalf = lane >> 4;
  const __bf16* p = buf + (size_t)col * kBStride + khalf * 8;
  Frag f;
  f.u4[0] = *reinterpret_cast<const u32x4*>(p);
  f.u4[1] = *reinterpret_cast<const u32x4*>(p + 16);
  return f.v;
}

// Stage one 256-col x 32-K bf16 weight tile into LDS asynchronously.
// 256 threads, 4 async b128 per thread (16KB total).
__device__ __forceinline__ void stage_b(const __bf16* __restrict__ Wt, int ldw,
                                        int kk, __bf16* buf, int tid) {
#pragma unroll
  for (int r = 0; r < 4; ++r) {
    int i = tid + r * 256;
    int c = i >> 2;       // column 0..255
    int q = i & 3;        // 16B chunk 0..3 within the 64B row segment
    async_g2l_b128((unsigned)(uintptr_t)(buf + (size_t)c * kBStride + q * 8),
                   Wt + (size_t)c * ldw + kk + q * 8);
  }
}

// Double-buffered, async-staged block GEMM accumulation over K.
// A: wave-local (already offset to wave m0), W: block-local (offset to block n0).
__device__ __forceinline__ void gemm_staged(const __bf16* __restrict__ A, int lda,
                                            const __bf16* __restrict__ W, int ldw,
                                            int K, __bf16* sm, v8f (&acc)[2][4],
                                            int tid, int lane, int wn) {
  int cur = 0;
  stage_b(W, ldw, 0, sm, tid);
  wait_async0();
  __syncthreads();
  for (int kk = 0; kk < K; kk += 32) {
    if (kk + 32 < K)  // overlap next-tile async copy with this tile's WMMAs
      stage_b(W, ldw, kk + 32, sm + (size_t)(cur ^ 1) * kBufElems, tid);

    v16bf a0 = load_a_frag(A, lda, kk, lane);
    v16bf a1 = load_a_frag(A + (size_t)16 * lda, lda, kk, lane);
    const __bf16* bb = sm + (size_t)cur * kBufElems + (size_t)(wn * 64) * kBStride;
#pragma unroll
    for (int ni = 0; ni < 4; ++ni) {
      v16bf bfrag = load_b_frag_lds(bb + (size_t)ni * 16 * kBStride, lane);
      acc[0][ni] = wmma_bf16(a0, bfrag, acc[0][ni]);
      acc[1][ni] = wmma_bf16(a1, bfrag, acc[1][ni]);
    }
    wait_async0();
    __syncthreads();
    cur ^= 1;
  }
}

__device__ __forceinline__ void zero_acc(v8f (&acc)[2][4]) {
#pragma unroll
  for (int mi = 0; mi < 2; ++mi)
#pragma unroll
    for (int ni = 0; ni < 4; ++ni)
#pragma unroll
      for (int r = 0; r < 8; ++r) acc[mi][ni][r] = 0.0f;
}

// ---------------- GEMM 1: feat = relu(x @ W1^T + b1), bf16 out --------------
__global__ void __launch_bounds__(256)
gemm_relu_bf16(const __bf16* __restrict__ A, const __bf16* __restrict__ W,
               const float* __restrict__ bias, __bf16* __restrict__ C,
               int M, int N, int K, int lda, int ldw) {
  __shared__ __bf16 sm[2 * kBufElems];
  int tid  = threadIdx.x;
  int lane = tid & 31;
  int w    = tid >> 5;
  int wm = w >> 2, wn = w & 3;
  int m0 = blockIdx.y * 64 + wm * 32;
  int nblk = blockIdx.x * 256;

  v8f acc[2][4];
  zero_acc(acc);
  gemm_staged(A + (size_t)m0 * lda, lda, W + (size_t)nblk * ldw, ldw, K,
              sm, acc, tid, lane, wn);

  int nlane = lane & 15, mh = lane >> 4;
#pragma unroll
  for (int mi = 0; mi < 2; ++mi)
#pragma unroll
    for (int ni = 0; ni < 4; ++ni) {
      int ncol = nblk + wn * 64 + ni * 16 + nlane;
      float bv = bias[ncol];
#pragma unroll
      for (int r = 0; r < 8; ++r) {
        int mrow = m0 + mi * 16 + mh * 8 + r;  // C layout: VGPR r -> M=r / r+8
        float v = acc[mi][ni][r] + bv;
        v = fmaxf(v, 0.0f);
        C[(size_t)mrow * N + ncol] = (__bf16)v;
      }
    }
}

// -------- per-step fused 4-direction gate GEMM: g = x_t@Wih^T + h@Whh^T + b --
__global__ void __launch_bounds__(256)
lstm_step_gemm(const __bf16* __restrict__ featb, const __bf16* __restrict__ featTb,
               const __bf16* __restrict__ hb, const __bf16* __restrict__ Wihb,
               const __bf16* __restrict__ Whhb, const float* __restrict__ bsum,
               float* __restrict__ g, int t) {
  __shared__ __bf16 sm[2 * kBufElems];
  int dir  = blockIdx.z;                 // 0=left 1=right 2=up 3=down
  int widx = (dir <= 1) ? dir : 2;       // up/down share the "d" weights

  const __bf16* A1;
  if (dir == 0)      A1 = featb  + (size_t)t * kD;
  else if (dir == 1) A1 = featb  + (size_t)(kS - 1 - t) * kD;
  else if (dir == 2) A1 = featTb + (size_t)t * kS;
  else               A1 = featTb + (size_t)(kD - 1 - t) * kS;
  const __bf16* A2 = hb + (size_t)dir * kB * kH;

  const __bf16* W1   = Wihb + (size_t)widx * kG * kD;
  const __bf16* W2   = Whhb + (size_t)widx * kG * kH;
  const float*  bias = bsum + (size_t)widx * kG;
  float*        out  = g    + (size_t)dir  * kB * kG;

  int tid  = threadIdx.x;
  int lane = tid & 31;
  int w    = tid >> 5;
  int wm = w >> 2, wn = w & 3;
  int m0 = blockIdx.y * 64 + wm * 32;
  int nblk = blockIdx.x * 256;

  v8f acc[2][4];
  zero_acc(acc);
  // segment 1: x_t @ Wih^T (K=64)
  gemm_staged(A1 + (size_t)m0 * kFS, kFS, W1 + (size_t)nblk * kD, kD, kD,
              sm, acc, tid, lane, wn);
  // segment 2: h @ Whh^T (K=512)
  gemm_staged(A2 + (size_t)m0 * kH, kH, W2 + (size_t)nblk * kH, kH, kH,
              sm, acc, tid, lane, wn);

  int nlane = lane & 15, mh = lane >> 4;
#pragma unroll
  for (int mi = 0; mi < 2; ++mi)
#pragma unroll
    for (int ni = 0; ni < 4; ++ni) {
      int ncol = nblk + wn * 64 + ni * 16 + nlane;
      float bv = bias[ncol];
#pragma unroll
      for (int r = 0; r < 8; ++r) {
        int mrow = m0 + mi * 16 + mh * 8 + r;
        out[(size_t)mrow * kG + ncol] = acc[mi][ni][r] + bv;
      }
    }
}

// ---------------- pointwise LSTM gate update ----------------
__device__ __forceinline__ float sigf(float x) { return 1.0f / (1.0f + __expf(-x)); }

__global__ void __launch_bounds__(256)
lstm_pointwise(const float* __restrict__ g, float* __restrict__ c,
               __bf16* __restrict__ hb) {
  size_t idx = (size_t)blockIdx.x * blockDim.x + threadIdx.x;  // [4][B][H]
  if (idx >= (size_t)4 * kB * kH) return;
  int dir = (int)(idx / ((size_t)kB * kH));
  size_t rem = idx - (size_t)dir * kB * kH;
  int b  = (int)(rem / kH);
  int hh = (int)(rem % kH);
  const float* gr = g + (size_t)dir * kB * kG + (size_t)b * kG;
  float gi = sigf(gr[hh]);
  float gf = sigf(gr[kH + hh]);
  float gg = tanhf(gr[2 * kH + hh]);
  float go = sigf(gr[3 * kH + hh]);
  float cn = gf * c[idx] + gi * gg;
  c[idx]  = cn;
  hb[idx] = (__bf16)(go * tanhf(cn));
}

// -------- 64x64 inner-tile transpose (for up/down directions), via LDS ------
__global__ void __launch_bounds__(256)
transpose64(const __bf16* __restrict__ src, __bf16* __restrict__ dst) {
  int b = blockIdx.x;
  const __bf16* s = src + (size_t)b * (kS * kD);
  __bf16* d       = dst + (size_t)b * (kS * kD);
  __shared__ __bf16 tile[64][65];
  for (int i = threadIdx.x; i < 4096; i += 256) tile[i >> 6][i & 63] = s[i];
  __syncthreads();
  for (int i = threadIdx.x; i < 4096; i += 256) d[i] = tile[i & 63][i >> 6];
}

// ---------------- small helpers ----------------
__global__ void cvt_f32_bf16(const float* __restrict__ src, __bf16* __restrict__ dst,
                             long n) {
  long i = (long)blockIdx.x * blockDim.x + threadIdx.x;
  if (i < n) dst[i] = (__bf16)src[i];
}

__global__ void bias_sum(const float* __restrict__ a, const float* __restrict__ b,
                         float* __restrict__ dst, int n) {
  int i = blockIdx.x * blockDim.x + threadIdx.x;
  if (i < n) dst[i] = a[i] + b[i];
}

// ---------------- logits + log_softmax ----------------
__global__ void __launch_bounds__(32)
logits_lsm(const __bf16* __restrict__ hb, const float* __restrict__ W3,
           const float* __restrict__ b3, float* __restrict__ out) {
  int b = blockIdx.x;
  int lane = threadIdx.x;
  float acc[10];
#pragma unroll
  for (int j = 0; j < 10; ++j) acc[j] = 0.0f;
  for (int k = lane; k < kG; k += 32) {
    int dir = k >> 9, hh = k & (kH - 1);
    float xv = (float)hb[(size_t)dir * kB * kH + (size_t)b * kH + hh];
#pragma unroll
    for (int j = 0; j < 10; ++j) acc[j] += xv * W3[j * kG + k];
  }
#pragma unroll
  for (int off = 16; off > 0; off >>= 1)
#pragma unroll
    for (int j = 0; j < 10; ++j) acc[j] += __shfl_down(acc[j], off, 32);
  if (lane == 0) {
    float lg[10], mx = -1e30f;
#pragma unroll
    for (int j = 0; j < 10; ++j) { lg[j] = acc[j] + b3[j]; mx = fmaxf(mx, lg[j]); }
    float s = 0.0f;
#pragma unroll
    for (int j = 0; j < 10; ++j) s += __expf(lg[j] - mx);
    float lse = mx + __logf(s);
#pragma unroll
    for (int j = 0; j < 10; ++j) out[(size_t)b * 10 + j] = lg[j] - lse;
  }
}

// ---------------- host orchestration ----------------
extern "C" void kernel_launch(void* const* d_in, const int* in_sizes, int n_in,
                              void* d_out, int out_size, void* d_ws, size_t ws_size,
                              hipStream_t stream) {
  (void)in_sizes; (void)n_in; (void)out_size; (void)ws_size;
  const float* x  = (const float*)d_in[0];
  const float* h0 = (const float*)d_in[1];
  const float* c0 = (const float*)d_in[2];
  const float* W1 = (const float*)d_in[3];
  const float* b1 = (const float*)d_in[4];
  const float* Wih[3] = { (const float*)d_in[5], (const float*)d_in[9],  (const float*)d_in[13] };
  const float* Whh[3] = { (const float*)d_in[6], (const float*)d_in[10], (const float*)d_in[14] };
  const float* bih[3] = { (const float*)d_in[7], (const float*)d_in[11], (const float*)d_in[15] };
  const float* bhh[3] = { (const float*)d_in[8], (const float*)d_in[12], (const float*)d_in[16] };
  const float* W3 = (const float*)d_in[17];
  const float* b3 = (const float*)d_in[18];
  float* out = (float*)d_out;

  char* ws = (char*)d_ws;
  size_t off = 0;
  auto alloc = [&](size_t bytes) -> void* {
    off = (off + 255) & ~(size_t)255;
    void* p = ws + off;
    off += bytes;
    return p;
  };

  __bf16* xb     = (__bf16*)alloc((size_t)kB * kIN * 2);
  __bf16* W1b    = (__bf16*)alloc((size_t)kFS * kIN * 2);
  __bf16* featb  = (__bf16*)alloc((size_t)kB * kFS * 2);
  __bf16* featTb = (__bf16*)alloc((size_t)kB * kFS * 2);
  __bf16* Wihb   = (__bf16*)alloc((size_t)3 * kG * kD * 2);
  __bf16* Whhb   = (__bf16*)alloc((size_t)3 * kG * kH * 2);
  float*  bsum   = (float*) alloc((size_t)3 * kG * 4);
  __bf16* hb     = (__bf16*)alloc((size_t)4 * kB * kH * 2);
  float*  cst    = (float*) alloc((size_t)4 * kB * kH * 4);
  float*  g      = (float*) alloc((size_t)4 * kB * kG * 4);

  const int T = 256;
  auto blocks = [](long n, int t) { return (unsigned)((n + t - 1) / t); };

  // bf16 conversions of activations / weights
  cvt_f32_bf16<<<blocks((long)kB * kIN, T), T, 0, stream>>>(x, xb, (long)kB * kIN);
  cvt_f32_bf16<<<blocks((long)kFS * kIN, T), T, 0, stream>>>(W1, W1b, (long)kFS * kIN);
  for (int w = 0; w < 3; ++w) {
    cvt_f32_bf16<<<blocks((long)kG * kD, T), T, 0, stream>>>(
        Wih[w], Wihb + (size_t)w * kG * kD, (long)kG * kD);
    cvt_f32_bf16<<<blocks((long)kG * kH, T), T, 0, stream>>>(
        Whh[w], Whhb + (size_t)w * kG * kH, (long)kG * kH);
    bias_sum<<<blocks(kG, T), T, 0, stream>>>(bih[w], bhh[w], bsum + (size_t)w * kG, kG);
  }
  cvt_f32_bf16<<<blocks((long)4 * kB * kH, T), T, 0, stream>>>(h0, hb, (long)4 * kB * kH);
  hipMemcpyAsync(cst, c0, (size_t)4 * kB * kH * 4, hipMemcpyDeviceToDevice, stream);

  // feat = relu(x @ W1^T + b1)
  gemm_relu_bf16<<<dim3(kFS / 256, kB / 64), 256, 0, stream>>>(
      xb, W1b, b1, featb, kB, kFS, kIN, kIN, kIN);

  // transpose inner 64x64 tile for up/down directions
  transpose64<<<kB, 256, 0, stream>>>(featb, featTb);

  // 64 recurrent steps: fused 4-direction WMMA gate GEMM + pointwise update
  for (int t = 0; t < kS; ++t) {
    lstm_step_gemm<<<dim3(kG / 256, kB / 64, 4), 256, 0, stream>>>(
        featb, featTb, hb, Wihb, Whhb, bsum, g, t);
    lstm_pointwise<<<blocks((long)4 * kB * kH, T), T, 0, stream>>>(g, cst, hb);
  }

  // classifier + log_softmax
  logits_lsm<<<kB, 32, 0, stream>>>(hb, W3, b3, out);
}
